// RegionProposalNetwork_66357244723882
// MI455X (gfx1250) — matrix operand order repaired and checked by reading, not compiled, not run on hardware
//
#include <hip/hip_runtime.h>

typedef __attribute__((ext_vector_type(16))) _Float16 v16h;
typedef __attribute__((ext_vector_type(8)))  _Float16 v8h;
typedef __attribute__((ext_vector_type(8)))  float    v8f;
typedef int i32x4v __attribute__((vector_size(16)));   // matches builtin's V4i

#define HH 200
#define WW 200
#define CC 512
#define AA 9
#define NPIX 40000
#define NANCH 360000
#define KDIM 4608       // 512 * 9
#define NKC (KDIM / 32) // 144 K-chunks
#define KPRE 1000
#define KPOST 100
#define IMGSZ 3200.0f
#define IOU_TH 0.7f

#define TILEK 32
#define MT 128
#define NT 128

union F16x16 { v16h v; v8h h[2]; };

#if __has_builtin(__builtin_amdgcn_global_load_async_to_lds_b128)
#define USE_ASYNC 1
#else
#define USE_ASYNC 0
#endif

#if __has_builtin(__builtin_amdgcn_s_wait_asynccnt)
#define WAIT_ASYNCCNT(n) __builtin_amdgcn_s_wait_asynccnt(n)
#else
#define WAIT_ASYNCCNT(n) asm volatile("s_wait_asynccnt %0" :: "i"(n) : "memory")
#endif

__device__ __forceinline__ void async_cp16(const _Float16* g, _Float16* l) {
#if USE_ASYNC
  __builtin_amdgcn_global_load_async_to_lds_b128(
      (__attribute__((address_space(1))) i32x4v*)(g),
      (__attribute__((address_space(3))) i32x4v*)(l), 0, 0);
#endif
}

// ---------------------------------------------------------------------------
// Packing kernels: f32 NCHW -> f16 layouts friendly to WMMA implicit GEMM
// ---------------------------------------------------------------------------

// feat (1,512,200,200) f32 NCHW -> featN[pix][c] f16 (NHWC)
__global__ __launch_bounds__(256) void k_pack_feat(const float* __restrict__ feat,
                                                   _Float16* __restrict__ featN) {
  int t = blockIdx.x * 256 + threadIdx.x;          // 20,480,000 exact
  int c = t & 511;
  int pix = t >> 9;
  featN[t] = (_Float16)feat[(size_t)c * NPIX + pix];
}

// conv_w (512,512,3,3) -> Wp[n][k] f16, k = (dy*3+dx)*512 + c
__global__ __launch_bounds__(256) void k_pack_w(const float* __restrict__ w,
                                                _Float16* __restrict__ Wp) {
  int t = blockIdx.x * 256 + threadIdx.x;          // 2,359,296 exact
  int n = t / KDIM;
  int k = t - n * KDIM;
  int plane = k >> 9;                               // dy*3+dx
  int c = k & 511;
  Wp[t] = (_Float16)w[(size_t)(n * 512 + c) * 9 + plane];
}

// cls_w (9,512) + reg_w (36,512) -> W2[48][512] f16 (rows 45..47 zero)
__global__ __launch_bounds__(256) void k_pack_w2(const float* __restrict__ cw,
                                                 const float* __restrict__ rw,
                                                 _Float16* __restrict__ W2) {
  int t = blockIdx.x * 256 + threadIdx.x;          // 24,576 exact
  int n = t >> 9;
  int c = t & 511;
  float v = 0.f;
  if (n < 9) v = cw[n * 512 + c];
  else if (n < 45) v = rw[(n - 9) * 512 + c];
  W2[t] = (_Float16)v;
}

__global__ __launch_bounds__(256) void k_zero_u32(unsigned* __restrict__ p, int n) {
  int t = blockIdx.x * 256 + threadIdx.x;
  if (t < n) p[t] = 0u;
}

// ---------------------------------------------------------------------------
// GEMM1: 3x3 conv as implicit GEMM  M=40000, N=512, K=4608, f16 WMMA, f32 acc
// block 256 thr = 8 wave32; tile 128(M)x128(N); wave strip 32x64 = 8 WMMA/step
// triple-buffered LDS fed by GLOBAL_LOAD_ASYNC_TO_LDS_B128 (2 tiles in flight)
// ---------------------------------------------------------------------------
__global__ __launch_bounds__(256) void k_gemm1(const _Float16* __restrict__ featN,
                                               const _Float16* __restrict__ Wp,
                                               const float* __restrict__ convb,
                                               const _Float16* __restrict__ zpad,
                                               _Float16* __restrict__ xh) {
#if USE_ASYNC
  __shared__ _Float16 lsA[3][MT * TILEK];
  __shared__ _Float16 lsB[3][NT * TILEK];
#else
  __shared__ _Float16 lsA[1][MT * TILEK];
  __shared__ _Float16 lsB[1][NT * TILEK];
#endif

  int mt = blockIdx.x >> 2;            // 0..312
  int nt = blockIdx.x & 3;             // 0..3
  int p0 = mt * MT;
  int n0 = nt * NT;

  int tid = threadIdx.x;
  int wave = tid >> 5, lane = tid & 31;
  int wm = wave & 3, wn = wave >> 2;
  int lm = lane & 15, half = lane >> 4;

  // staging: thread -> row (tid>>1), 32B half-row chunk pair at (tid&1)*2
  int srow = tid >> 1;
  int scp = (tid & 1) * 2;             // 8-half chunk index: scp, scp+1
  int apix = p0 + srow;
  bool prow_ok = apix < NPIX;
  int apc = prow_ok ? apix : 0;
  int ay = apc / WW, ax = apc - ay * WW;

  v8f acc[2][4] = {};

  auto issue = [&](int kc, int buf) {
    int plane = kc >> 4;               // uniform scalar: (kc*32)/512
    int c0 = (kc & 15) << 5;
    int dy = plane / 3 - 1;            // SALU magic div, no scratch arrays
    int dx = plane - (plane / 3) * 3 - 1;
    int yy = ay + dy, xx = ax + dx;
    bool ok = prow_ok && (unsigned)yy < (unsigned)HH && (unsigned)xx < (unsigned)WW;
    const _Float16* ga =
        ok ? (featN + (((size_t)(yy * WW + xx)) << 9) + c0 + scp * 8) : zpad;
    const _Float16* ga2 = ok ? ga + 8 : zpad;
    _Float16* la = &lsA[buf][srow * TILEK + scp * 8];
    async_cp16(ga, la);
    async_cp16(ga2, la + 8);
    const _Float16* gb = Wp + (size_t)(n0 + srow) * KDIM + kc * TILEK + scp * 8;
    _Float16* lb = &lsB[buf][srow * TILEK + scp * 8];
    async_cp16(gb, lb);
    async_cp16(gb + 8, lb + 8);
  };

  auto compute = [&](int buf) {
    F16x16 af[2];
#pragma unroll
    for (int mi = 0; mi < 2; ++mi) {
      const _Float16* ap = &lsA[buf][(wm * 32 + mi * 16 + lm) * TILEK + half * 8];
      af[mi].h[0] = *(const v8h*)ap;
      af[mi].h[1] = *(const v8h*)(ap + 16);
    }
#pragma unroll
    for (int j = 0; j < 4; ++j) {
      F16x16 bf;
      const _Float16* bp = &lsB[buf][(wn * 64 + j * 16 + lm) * TILEK + half * 8];
      bf.h[0] = *(const v8h*)bp;
      bf.h[1] = *(const v8h*)(bp + 16);
#pragma unroll
      for (int mi = 0; mi < 2; ++mi)
        acc[mi][j] = __builtin_amdgcn_wmma_f32_16x16x32_f16(
            false, af[mi].v, false, bf.v, (short)0, acc[mi][j], false, false);
    }
  };

#if USE_ASYNC
  issue(0, 0);
  issue(1, 1);
#pragma unroll 3
  for (int kc = 0; kc < NKC; ++kc) {
    if (kc + 2 < NKC) {
      issue(kc + 2, (kc + 2) % 3);
      WAIT_ASYNCCNT(8);      // tile kc's 4 async loads drained (in-order)
    } else if (kc + 1 < NKC) {
      WAIT_ASYNCCNT(4);
    } else {
      WAIT_ASYNCCNT(0);
    }
    __syncthreads();
    compute(kc % 3);
    __syncthreads();
  }
#else
  for (int kc = 0; kc < NKC; ++kc) {
    int plane = kc >> 4;
    int c0 = (kc & 15) << 5;
    int dy = plane / 3 - 1;
    int dx = plane - (plane / 3) * 3 - 1;
    int yy = ay + dy, xx = ax + dx;
    bool ok = prow_ok && (unsigned)yy < (unsigned)HH && (unsigned)xx < (unsigned)WW;
    uint4 a0 = {0u, 0u, 0u, 0u}, a1 = {0u, 0u, 0u, 0u};
    if (ok) {
      const uint4* g =
          (const uint4*)(featN + (((size_t)(yy * WW + xx)) << 9) + c0 + scp * 8);
      a0 = g[0];
      a1 = g[1];
    }
    const uint4* gb =
        (const uint4*)(Wp + (size_t)(n0 + srow) * KDIM + kc * TILEK + scp * 8);
    uint4 b0 = gb[0], b1 = gb[1];
    __syncthreads();
    *(uint4*)(&lsA[0][srow * TILEK + scp * 8]) = a0;
    *(uint4*)(&lsA[0][srow * TILEK + scp * 8 + 8]) = a1;
    *(uint4*)(&lsB[0][srow * TILEK + scp * 8]) = b0;
    *(uint4*)(&lsB[0][srow * TILEK + scp * 8 + 8]) = b1;
    __syncthreads();
    compute(0);
  }
#endif

  // bias + ReLU, store NHWC f16
#pragma unroll
  for (int j = 0; j < 4; ++j) {
    int cout = n0 + wn * 64 + j * 16 + lm;
    float bias = convb[cout];
#pragma unroll
    for (int mi = 0; mi < 2; ++mi)
#pragma unroll
      for (int r = 0; r < 8; ++r) {
        int pix = p0 + wm * 32 + mi * 16 + half * 8 + r;
        if (pix < NPIX) {
          float v = acc[mi][j][r] + bias;
          v = v > 0.f ? v : 0.f;
          xh[((size_t)pix << 9) + cout] = (_Float16)v;
        }
      }
  }
}

// ---------------------------------------------------------------------------
// GEMM2: 1x1 heads  M=40000, N=48 (9 cls + 36 reg + 3 pad), K=512
// whole B (48x512 f16 = 48KB) resident in LDS; A streamed from global
// ---------------------------------------------------------------------------
__global__ __launch_bounds__(256) void k_gemm2(const _Float16* __restrict__ xh,
                                               const _Float16* __restrict__ W2,
                                               const float* __restrict__ clsb,
                                               const float* __restrict__ regb,
                                               float* __restrict__ cls,
                                               float* __restrict__ deltas) {
  __shared__ _Float16 sW[48 * 512];
  int tid = threadIdx.x;
#pragma unroll
  for (int i = 0; i < 12; ++i)
    *(uint4*)(&sW[tid * 96 + i * 8]) = *(const uint4*)(W2 + tid * 96 + i * 8);
  __syncthreads();

  int wave = tid >> 5, lane = tid & 31, lm = lane & 15, half = lane >> 4;
  int p0 = blockIdx.x * 128;
  int prow = p0 + wave * 16 + lm;
  if (prow > NPIX - 1) prow = NPIX - 1;

  v8f acc[3] = {};
  for (int kc = 0; kc < 16; ++kc) {
    F16x16 af;
    const _Float16* ap = xh + ((size_t)prow << 9) + kc * 32 + half * 8;
    af.h[0] = *(const v8h*)ap;
    af.h[1] = *(const v8h*)(ap + 16);
#pragma unroll
    for (int j = 0; j < 3; ++j) {
      F16x16 bf;
      const _Float16* bp = &sW[(j * 16 + lm) * 512 + kc * 32 + half * 8];
      bf.h[0] = *(const v8h*)bp;
      bf.h[1] = *(const v8h*)(bp + 16);
      acc[j] = __builtin_amdgcn_wmma_f32_16x16x32_f16(false, af.v, false, bf.v,
                                                      (short)0, acc[j], false, false);
    }
  }

#pragma unroll
  for (int j = 0; j < 3; ++j) {
    int n2 = j * 16 + lm;
#pragma unroll
    for (int r = 0; r < 8; ++r) {
      int pix = p0 + wave * 16 + half * 8 + r;
      if (pix >= NPIX) continue;
      float v = acc[j][r];
      if (n2 < 9) cls[pix * 9 + n2] = v + clsb[n2];
      else if (n2 < 45) { int d = n2 - 9; deltas[pix * 36 + d] = v + regb[d]; }
    }
  }
}

// ---------------------------------------------------------------------------
// Decode anchors + sigmoid
// ---------------------------------------------------------------------------
__global__ __launch_bounds__(256) void k_decode(const float* __restrict__ cls,
                                                const float* __restrict__ deltas,
                                                float* __restrict__ boxes,
                                                float* __restrict__ scores) {
  int i = blockIdx.x * 256 + threadIdx.x;
  if (i >= NANCH) return;
  int a = i % AA;
  int pix = i / AA;
  int y = pix / WW, x = pix - y * WW;
  int ri = a / 3, si = a - ri * 3;
  float scl = (si == 0) ? 128.f : ((si == 1) ? 256.f : 512.f);
  float hr  = (ri == 0) ? 0.70710678118f : ((ri == 1) ? 1.0f : 1.41421356237f);
  float wr = 1.0f / hr;
  float wa = wr * scl, ha = hr * scl;
  float cx = x * 16.0f, cy = y * 16.0f;

  float4 d = ((const float4*)deltas)[i];
  float px = d.x * wa + cx;
  float py = d.y * ha + cy;
  float pw = expf(d.z) * wa;
  float ph = expf(d.w) * ha;
  float4 b;
  b.x = px - 0.5f * pw; b.y = py - 0.5f * ph;
  b.z = px + 0.5f * pw; b.w = py + 0.5f * ph;
  ((float4*)boxes)[i] = b;

  float l = cls[i];
  scores[i] = 1.0f / (1.0f + expf(-l));
}

// ---------------------------------------------------------------------------
// Top-1000: two-level 16-bit radix threshold select (exact 32-bit threshold)
// ---------------------------------------------------------------------------
__global__ __launch_bounds__(256) void k_hist1(const float* __restrict__ scores,
                                               unsigned* __restrict__ hist1) {
  int i = blockIdx.x * 256 + threadIdx.x;
  if (i >= NANCH) return;
  unsigned key = __float_as_uint(scores[i]) >> 16;   // scores in (0,1): monotonic
  atomicAdd(&hist1[key], 1u);
}

__global__ void k_findT(const unsigned* __restrict__ hist1, unsigned* __restrict__ meta) {
  if (threadIdx.x != 0) return;
  unsigned cum = 0, T = 0;
  for (int b = 65535; b >= 0; --b) {
    unsigned h = hist1[b];
    if (cum + h >= KPRE) { T = (unsigned)b; break; }
    cum += h;
  }
  meta[0] = T;    // bucket holding the K-th value
  meta[1] = cum;  // count strictly above bucket T
}

__global__ __launch_bounds__(256) void k_hist2(const float* __restrict__ scores,
                                               const unsigned* __restrict__ meta,
                                               unsigned* __restrict__ hist2) {
  int i = blockIdx.x * 256 + threadIdx.x;
  if (i >= NANCH) return;
  unsigned bits = __float_as_uint(scores[i]);
  if ((bits >> 16) == meta[0]) atomicAdd(&hist2[bits & 0xFFFFu], 1u);
}

__global__ void k_findT2(const unsigned* __restrict__ hist2, unsigned* __restrict__ meta) {
  if (threadIdx.x != 0) return;
  unsigned cum = meta[1], L = 0;
  for (int b = 65535; b >= 0; --b) {
    unsigned h = hist2[b];
    if (cum + h >= KPRE) { L = (unsigned)b; break; }
    cum += h;
  }
  meta[2] = (meta[0] << 16) | L;  // exact 32-bit threshold bits
  meta[3] = cum;                  // count strictly above exact threshold
}

__global__ __launch_bounds__(256) void k_compact(const float* __restrict__ scores,
                                                 unsigned* __restrict__ meta,
                                                 float* __restrict__ tks,
                                                 int* __restrict__ tki) {
  int i = blockIdx.x * 256 + threadIdx.x;
  if (i >= NANCH) return;
  float s = scores[i];
  unsigned bits = __float_as_uint(s);
  unsigned thr = meta[2];
  if (bits > thr) {
    unsigned slot = atomicAdd(&meta[4], 1u);
    tks[slot] = s; tki[slot] = i;
  } else if (bits == thr) {
    unsigned t = atomicAdd(&meta[5], 1u);
    unsigned slot = meta[3] + t;
    if (slot < KPRE) { tks[slot] = s; tki[slot] = i; }
  }
}

// Bitonic sort 1024 entries (1000 real + pad) desc by score, asc by index on ties,
// then gather + clamp boxes for NMS.
__global__ __launch_bounds__(512) void k_sort(const float* __restrict__ tks,
                                              const int* __restrict__ tki,
                                              const float* __restrict__ boxes,
                                              float* __restrict__ props,
                                              float* __restrict__ sc) {
  __shared__ float ss[1024];
  __shared__ int   si[1024];
  int t = threadIdx.x;
  for (int i = t; i < 1024; i += 512) {
    if (i < KPRE) { ss[i] = tks[i]; si[i] = tki[i]; }
    else          { ss[i] = -1.0f;  si[i] = 0x7FFFFFFF; }
  }
  __syncthreads();
  for (int k = 2; k <= 1024; k <<= 1) {
    for (int j = k >> 1; j > 0; j >>= 1) {
      for (int i = t; i < 1024; i += 512) {
        int ixj = i ^ j;
        if (ixj > i) {
          bool up = ((i & k) == 0);  // "up" region = descending overall
          bool good = (ss[i] > ss[ixj]) ||
                      (ss[i] == ss[ixj] && si[i] < si[ixj]);
          if (up ? !good : good) {
            float fs = ss[i]; ss[i] = ss[ixj]; ss[ixj] = fs;
            int   fi = si[i]; si[i] = si[ixj]; si[ixj] = fi;
          }
        }
      }
      __syncthreads();
    }
  }
  for (int i = t; i < KPRE; i += 512) {
    int ii = si[i];
    float4 b = ((const float4*)boxes)[ii];
    b.x = fminf(fmaxf(b.x, 0.f), IMGSZ);
    b.y = fminf(fmaxf(b.y, 0.f), IMGSZ);
    b.z = fminf(fmaxf(b.z, 0.f), IMGSZ);
    b.w = fminf(fmaxf(b.w, 0.f), IMGSZ);
    ((float4*)props)[i] = b;
    sc[i] = ss[i];
  }
}

// ---------------------------------------------------------------------------
// Greedy NMS over 1000 sorted boxes + final compaction to 100 outputs
// ---------------------------------------------------------------------------
__global__ __launch_bounds__(1024) void k_nms(const float* __restrict__ props,
                                              const float* __restrict__ sc,
                                              float* __restrict__ out) {
  __shared__ float x1[KPRE], y1[KPRE], x2[KPRE], y2[KPRE], ar[KPRE];
  __shared__ unsigned char kp[KPRE];
  __shared__ int pos[KPRE];
  int t = threadIdx.x;

  for (int i = t; i < 500; i += 1024) out[i] = 0.0f;  // zero out_boxes + out_scores
  for (int i = t; i < KPRE; i += 1024) {
    float4 b = ((const float4*)props)[i];
    x1[i] = b.x; y1[i] = b.y; x2[i] = b.z; y2[i] = b.w;
    ar[i] = (b.z - b.x) * (b.w - b.y);
    kp[i] = 1;
  }
  __syncthreads();

  for (int i = 0; i < KPRE; ++i) {
    __syncthreads();
    if (!kp[i]) continue;
    float xi1 = x1[i], yi1 = y1[i], xi2 = x2[i], yi2 = y2[i], ai = ar[i];
    for (int j = i + 1 + t; j < KPRE; j += 1024) {
      if (!kp[j]) continue;
      float iw = fminf(xi2, x2[j]) - fmaxf(xi1, x1[j]); iw = fmaxf(iw, 0.f);
      float ih = fminf(yi2, y2[j]) - fmaxf(yi1, y1[j]); ih = fmaxf(ih, 0.f);
      float inter = iw * ih;
      float iou = inter / (ai + ar[j] - inter);
      if (iou > IOU_TH) kp[j] = 0;
    }
  }
  __syncthreads();

  if (t == 0) {
    int c = 0;
    for (int i = 0; i < KPRE; ++i) { pos[i] = c; c += kp[i]; }
  }
  __syncthreads();

  for (int i = t; i < KPRE; i += 1024) {
    if (kp[i] && pos[i] < KPOST) {
      int p = pos[i];
      out[p * 4 + 0] = x1[i];
      out[p * 4 + 1] = y1[i];
      out[p * 4 + 2] = x2[i];
      out[p * 4 + 3] = y2[i];
      out[400 + p] = sc[i];
    }
  }
}

// ---------------------------------------------------------------------------
extern "C" void kernel_launch(void* const* d_in, const int* in_sizes, int n_in,
                              void* d_out, int out_size, void* d_ws, size_t ws_size,
                              hipStream_t stream) {
  const float* feat   = (const float*)d_in[0];
  const float* conv_w = (const float*)d_in[1];
  const float* conv_b = (const float*)d_in[2];
  const float* cls_w  = (const float*)d_in[3];
  const float* cls_b  = (const float*)d_in[4];
  const float* reg_w  = (const float*)d_in[5];
  const float* reg_b  = (const float*)d_in[6];

  char* ws = (char*)d_ws;
  size_t off = 0;
  auto alloc = [&](size_t bytes) -> char* {
    char* p = ws + off;
    off = (off + bytes + 255) & ~(size_t)255;
    return p;
  };

  _Float16* featN  = (_Float16*)alloc((size_t)NPIX * CC * 2);   // 41 MB NHWC f16
  _Float16* Wp     = (_Float16*)alloc((size_t)CC * KDIM * 2);   // 4.7 MB
  _Float16* W2     = (_Float16*)alloc((size_t)48 * 512 * 2);
  _Float16* xh     = (_Float16*)alloc((size_t)NPIX * CC * 2);   // conv out, f16 NHWC
  float*    cls    = (float*)alloc((size_t)NANCH * 4);
  float*    deltas = (float*)alloc((size_t)NANCH * 16);
  float*    boxes  = (float*)alloc((size_t)NANCH * 16);
  float*    scores = (float*)alloc((size_t)NANCH * 4);
  unsigned* hist1  = (unsigned*)alloc(65536 * 4);
  unsigned* hist2  = (unsigned*)alloc(65536 * 4);
  unsigned* meta   = (unsigned*)alloc(64 * 4);
  float*    tks    = (float*)alloc(1024 * 4);
  int*      tki    = (int*)alloc(1024 * 4);
  float*    props  = (float*)alloc((size_t)KPRE * 16);
  float*    sc     = (float*)alloc((size_t)KPRE * 4);
  _Float16* zpad   = (_Float16*)alloc(256);                      // 64B zero line (+pad)

  // pack + clear
  k_pack_feat<<<(NPIX * CC) / 256, 256, 0, stream>>>(feat, featN);
  k_pack_w<<<(CC * KDIM) / 256, 256, 0, stream>>>(conv_w, Wp);
  k_pack_w2<<<(48 * 512) / 256, 256, 0, stream>>>(cls_w, reg_w, W2);
  k_zero_u32<<<(65536 + 255) / 256, 256, 0, stream>>>(hist1, 65536);
  k_zero_u32<<<(65536 + 255) / 256, 256, 0, stream>>>(hist2, 65536);
  k_zero_u32<<<1, 64, 0, stream>>>(meta, 64);
  k_zero_u32<<<1, 64, 0, stream>>>((unsigned*)zpad, 64);

  // conv 3x3 (implicit GEMM) : 313 M-tiles * 4 N-tiles
  k_gemm1<<<313 * 4, 256, 0, stream>>>(featN, Wp, conv_b, zpad, xh);

  // heads
  k_gemm2<<<(NPIX + 127) / 128, 256, 0, stream>>>(xh, W2, cls_b, reg_b, cls, deltas);

  // decode + sigmoid
  int nb = (NANCH + 255) / 256;
  k_decode<<<nb, 256, 0, stream>>>(cls, deltas, boxes, scores);

  // top-1000 exact radix select
  k_hist1<<<nb, 256, 0, stream>>>(scores, hist1);
  k_findT<<<1, 64, 0, stream>>>(hist1, meta);
  k_hist2<<<nb, 256, 0, stream>>>(scores, meta, hist2);
  k_findT2<<<1, 64, 0, stream>>>(hist2, meta);
  k_compact<<<nb, 256, 0, stream>>>(scores, meta, tks, tki);
  k_sort<<<1, 512, 0, stream>>>(tks, tki, boxes, props, sc);

  // NMS + output
  k_nms<<<1, 1024, 0, stream>>>(props, sc, (float*)d_out);
}
